// GAT2017_75222057222852
// MI455X (gfx1250) — matrix operand order repaired and checked by reading, not compiled, not run on hardware
//
#include <hip/hip_runtime.h>
#include <hip/hip_bf16.h>

// ---------------------------------------------------------------------------
// 2-layer GAT for MI455X (gfx1250, wave32, WMMA).
// GEMMs: bf16 inputs, f32 accum via v_wmma_f32_16x16x32_bf16, with B panels
// double-buffered through LDS; k-loop manually unrolled x2 so LDS buffer
// offsets are compile-time and A fragments ping-pong without register copies.
// Edge phases: key-encoded atomicMax segment-max, exp+atomicAdd segment-sum,
// denominator folded into node-side finalize (one div per (node,head,c)).
// ---------------------------------------------------------------------------

#define NNODES   100000
#define INDIM    512
#define HEADS    4
#define C1       32
#define HC1      128     // HEADS*C1
#define C2       64
#define HC2      256     // HEADS*C2
#define NEG_SLOPE 0.2f

typedef __attribute__((ext_vector_type(16))) __bf16 v16bf;
typedef __attribute__((ext_vector_type(8)))  float  v8f;

union Frag {
  v16bf v;
  uint4 q[2];
  unsigned short u[16];
};

__device__ __forceinline__ unsigned short f2bf(float f) {
  unsigned u = __float_as_uint(f);
  u += 0x7FFFu + ((u >> 16) & 1u);          // round-to-nearest-even
  return (unsigned short)(u >> 16);
}
// Monotonic float -> uint key for atomicMax over signed floats.
__device__ __forceinline__ unsigned fkey(float f) {
  unsigned u = __float_as_uint(f);
  return (u & 0x80000000u) ? ~u : (u | 0x80000000u);
}
__device__ __forceinline__ float funkey(unsigned k) {
  unsigned u = (k & 0x80000000u) ? (k & 0x7FFFFFFFu) : ~k;
  return __uint_as_float(u);
}
__device__ __forceinline__ float lrelu(float x) {
  return x > 0.f ? x : NEG_SLOPE * x;
}

// ---------------- conversion / packing -------------------------------------

__global__ void cvt_bf16_kernel(const float* __restrict__ in,
                                unsigned short* __restrict__ out, long long n) {
  long long i = (long long)blockIdx.x * blockDim.x + threadIdx.x;
  if (i < n) out[i] = f2bf(in[i]);
}

// Pack W [K x Nout] f32 into bf16 WMMA B-fragments, fragment-major:
// frag f = kt*NT + nt ; per-lane 16 contiguous bf16 (32B) in fragment layout.
__global__ void pack_w_kernel(const float* __restrict__ W,
                              unsigned short* __restrict__ Wp, int K, int Nout) {
  int NT = Nout >> 4;
  int nfrag = (K >> 5) * NT;
  int t = blockIdx.x * blockDim.x + threadIdx.x;
  if (t >= nfrag * 32) return;
  int f = t >> 5, lane = t & 31;
  int kt = f / NT, nt = f - kt * NT;
  int col = nt * 16 + (lane & 15);
  int k0  = kt * 32 + ((lane & 16) ? 8 : 0);
  unsigned short* o = Wp + ((size_t)t << 4);
#pragma unroll
  for (int i = 0; i < 16; ++i)
    o[i] = f2bf(W[(size_t)(k0 + (i < 8 ? i : i + 8)) * Nout + col]);
}

// ---------------- bf16 WMMA GEMM with LDS-staged B --------------------------
// One pipeline step: prefetch A frag kt+1 + B slice kt+1 (global->regs),
// compute NT WMMAs from LDS buffer at curOff, stage prefetched slice at
// nxtOff, barrier. curOff/nxtOff are compile-time at every call site.
template <int NT, int KT, int SLICE_U16, int QPT>
__device__ __forceinline__ void gemm_step(
    const unsigned short* __restrict__ Bp,
    const unsigned short* __restrict__ aBase,
    unsigned short* sb, int tid, int lane, int kt, int curOff, int nxtOff,
    Frag& aCur, Frag& aNxt, v8f* acc) {
  uint4 r[QPT];
  const bool pf = (kt + 1 < KT);
  if (pf) {
    const unsigned short* ap = aBase + (kt + 1) * 32;
    aNxt.q[0] = *(const uint4*)(ap);
    aNxt.q[1] = *(const uint4*)(ap + 16);
    const uint4* gq = (const uint4*)(Bp + (size_t)(kt + 1) * SLICE_U16);
#pragma unroll
    for (int i = 0; i < QPT; ++i) r[i] = gq[tid + i * 256];
  }
  const unsigned short* sbase = sb + curOff;
#pragma unroll
  for (int nt = 0; nt < NT; ++nt) {
    Frag b;
    const unsigned short* bp = sbase + ((nt * 32 + lane) << 4);
    b.q[0] = *(const uint4*)(bp);
    b.q[1] = *(const uint4*)(bp + 16);
    acc[nt] = __builtin_amdgcn_wmma_f32_16x16x32_bf16(
        false, aCur.v, false, b.v, (short)0, acc[nt], false, false);
  }
  if (pf) {
    uint4* sq = (uint4*)(sb + nxtOff);
#pragma unroll
    for (int i = 0; i < QPT; ++i) sq[tid + i * 256] = r[i];
  }
  __syncthreads();
}

// D[M,NOUT] = A[M,K](bf16,row-major) * B(bf16, fragment-packed), f32 out.
// Block = 256 threads = 8 waves; each wave: 16 rows x NOUT cols.
template <int K, int NOUT>
__global__ __launch_bounds__(256) void gemm_bf16_kernel(
    const unsigned short* __restrict__ A,
    const unsigned short* __restrict__ Bp,
    float* __restrict__ D, int M) {
  constexpr int NT = NOUT / 16;            // B fragments per k-slice
  constexpr int KT = K / 32;               // k-slices (even: 16 or 4)
  constexpr int SLICE_U16 = NT * 512;      // ushorts per slice (NT*32*16)
  constexpr int SLICE_Q   = SLICE_U16 / 8; // uint4 per slice
  constexpr int QPT       = SLICE_Q / 256; // uint4 per thread per slice

  __shared__ __align__(16) unsigned short sb[2 * SLICE_U16];

  const int tid  = threadIdx.x;
  const int wave = tid >> 5;
  const int lane = tid & 31;
  int rowBase = blockIdx.x * 128 + wave * 16;
  const bool active = rowBase < M;
  if (rowBase > M - 16) rowBase = M - 16;  // clamp: M is a multiple of 16

  const int kHalf = (lane & 16) ? 8 : 0;   // A layout: lanes 16-31 hold K+8..
  const unsigned short* aBase =
      A + (size_t)(rowBase + (lane & 15)) * K + kHalf;

  v8f acc[NT] = {};

  // stage slice 0 into LDS buffer 0
  {
    const uint4* gq = (const uint4*)Bp;
    uint4* sq = (uint4*)sb;
#pragma unroll
    for (int i = 0; i < QPT; ++i) sq[tid + i * 256] = gq[tid + i * 256];
  }
  // preload A fragment 0
  Frag aF0, aF1;
  aF0.q[0] = *(const uint4*)(aBase);
  aF0.q[1] = *(const uint4*)(aBase + 16);
  __syncthreads();

  for (int kt = 0; kt < KT; kt += 2) {
    gemm_step<NT, KT, SLICE_U16, QPT>(Bp, aBase, sb, tid, lane, kt,
                                      0, SLICE_U16, aF0, aF1, acc);
    gemm_step<NT, KT, SLICE_U16, QPT>(Bp, aBase, sb, tid, lane, kt + 1,
                                      SLICE_U16, 0, aF1, aF0, acc);
  }

  if (active) {
    const int rOff = (lane & 16) ? 8 : 0;  // C/D layout
    const int col  = lane & 15;
#pragma unroll
    for (int nt = 0; nt < NT; ++nt) {
#pragma unroll
      for (int v = 0; v < 8; ++v)
        D[(size_t)(rowBase + v + rOff) * NOUT + nt * 16 + col] = acc[nt][v];
    }
  }
}

// ---------------- attention logits -----------------------------------------
// One thread per (node, head): a_src[n,h] = <h[n,h,:], att_src[h,:]> etc.
template <int C>
__global__ void attn_kernel(const float* __restrict__ H,
                            const float* __restrict__ att_src,
                            const float* __restrict__ att_dst,
                            float* __restrict__ as_, float* __restrict__ ad_,
                            int Nn) {
  int t = blockIdx.x * blockDim.x + threadIdx.x;
  if (t >= Nn * HEADS) return;
  int n = t >> 2, h = t & 3;
  const float* hp = H + (size_t)n * (HEADS * C) + h * C;
  const float* s  = att_src + h * C;
  const float* d  = att_dst + h * C;
  float sa = 0.f, sd = 0.f;
#pragma unroll 8
  for (int c = 0; c < C; ++c) { float v = hp[c]; sa += v * s[c]; sd += v * d[c]; }
  as_[t] = sa; ad_[t] = sd;
}

// ---------------- edge passes ----------------------------------------------

__global__ void edge_max_kernel(const int* __restrict__ src,
                                const int* __restrict__ dst,
                                const float* __restrict__ as_,
                                const float* __restrict__ ad_,
                                unsigned* __restrict__ mk, long long E4) {
  long long t = (long long)blockIdx.x * blockDim.x + threadIdx.x;
  if (t >= E4) return;
  long long e = t >> 2; int h = (int)(t & 3);
  int s = src[e], d = dst[e];
  float l = lrelu(as_[(size_t)s * 4 + h] + ad_[(size_t)d * 4 + h]);
  atomicMax(&mk[(size_t)d * 4 + h], fkey(l));
}

__global__ void edge_exp_kernel(const int* __restrict__ src,
                                const int* __restrict__ dst,
                                const float* __restrict__ as_,
                                const float* __restrict__ ad_,
                                const unsigned* __restrict__ mk,
                                float* __restrict__ ssum,
                                float* __restrict__ ew, long long E4) {
  long long t = (long long)blockIdx.x * blockDim.x + threadIdx.x;
  if (t >= E4) return;
  long long e = t >> 2; int h = (int)(t & 3);
  int s = src[e], d = dst[e];
  float l = lrelu(as_[(size_t)s * 4 + h] + ad_[(size_t)d * 4 + h]);
  float m = funkey(mk[(size_t)d * 4 + h]);
  float ex = __expf(l - m);
  ew[t] = ex;
  unsafeAtomicAdd(&ssum[(size_t)d * 4 + h], ex);
}

// Accumulate un-normalized messages: out[d,hc] += ew[e,h] * H[s,hc].
// Denominator applied in the finalize kernels (one div per node element).
template <int HC, int C>
__global__ void edge_agg_kernel(const int* __restrict__ src,
                                const int* __restrict__ dst,
                                const float* __restrict__ ew,
                                const float* __restrict__ H,
                                float* __restrict__ out, long long total) {
  long long t = (long long)blockIdx.x * blockDim.x + threadIdx.x;
  if (t >= total) return;
  long long e = t / HC; int hc = (int)(t - e * HC); int h = hc / C;
  int s = src[e], d = dst[e];
  float v = ew[e * 4 + h] * H[(size_t)s * HC + hc];
  unsafeAtomicAdd(&out[(size_t)d * HC + hc], v);
}

// ---------------- finalize -------------------------------------------------

// Layer 1: normalize, + b1, ELU, emit bf16 activations for the next GEMM.
__global__ void fin1_kernel(const float* __restrict__ agg,
                            const float* __restrict__ ssum,
                            const float* __restrict__ b1,
                            unsigned short* __restrict__ xb2, long long total) {
  long long t = (long long)blockIdx.x * blockDim.x + threadIdx.x;
  if (t >= total) return;
  long long n = t >> 7; int hc = (int)(t & 127); int h = hc >> 5;
  float v = agg[t] / ssum[n * 4 + h] + b1[hc];
  v = v > 0.f ? v : (__expf(v) - 1.f);     // ELU
  xb2[t] = f2bf(v);
}

// Layer 2: normalize, mean over heads, + b2 -> final output [N,64] f32.
__global__ void fin2_kernel(const float* __restrict__ agg,
                            const float* __restrict__ ssum,
                            const float* __restrict__ b2,
                            float* __restrict__ out, long long total) {
  long long t = (long long)blockIdx.x * blockDim.x + threadIdx.x;
  if (t >= total) return;
  long long n = t >> 6; int c = (int)(t & 63);
  float acc = 0.f;
#pragma unroll
  for (int h = 0; h < HEADS; ++h)
    acc += agg[n * HC2 + h * C2 + c] / ssum[n * 4 + h];
  out[t] = acc * 0.25f + b2[c];
}

// ---------------------------------------------------------------------------

static inline unsigned blocksFor(long long n, int bs) {
  return (unsigned)((n + bs - 1) / bs);
}

extern "C" void kernel_launch(void* const* d_in, const int* in_sizes, int n_in,
                              void* d_out, int out_size, void* d_ws, size_t ws_size,
                              hipStream_t stream) {
  const float* x   = (const float*)d_in[0];
  const int*   src = (const int*)d_in[1];
  const int*   dst = (const int*)d_in[2];
  const float* W1  = (const float*)d_in[3];
  const float* aS1 = (const float*)d_in[4];
  const float* aD1 = (const float*)d_in[5];
  const float* b1  = (const float*)d_in[6];
  const float* W2  = (const float*)d_in[7];
  const float* aS2 = (const float*)d_in[8];
  const float* aD2 = (const float*)d_in[9];
  const float* b2  = (const float*)d_in[10];

  const long long Nn = NNODES;
  const long long E  = in_sizes[1];

  // ---- carve workspace (256B-aligned slices) ----
  char* p = (char*)d_ws;
  auto carve = [&](size_t bytes) {
    char* r = p;
    p += (bytes + 255) & ~(size_t)255;
    return r;
  };
  unsigned short* XB  = (unsigned short*)carve(2ull * Nn * INDIM);   // x bf16
  unsigned short* W1P = (unsigned short*)carve(2ull * INDIM * HC1);  // packed
  unsigned short* W2P = (unsigned short*)carve(2ull * HC1 * HC2);    // packed
  float*    H1   = (float*)carve(4ull * Nn * HC1);
  float*    AS1  = (float*)carve(4ull * Nn * HEADS);
  float*    AD1  = (float*)carve(4ull * Nn * HEADS);
  unsigned* MK1  = (unsigned*)carve(4ull * Nn * HEADS);
  float*    S1   = (float*)carve(4ull * Nn * HEADS);
  float*    EW1  = (float*)carve(4ull * E * HEADS);
  float*    AGG1 = (float*)carve(4ull * Nn * HC1);
  unsigned short* XB2 = (unsigned short*)carve(2ull * Nn * HC1);
  float*    H2   = (float*)carve(4ull * Nn * HC2);
  float*    AS2  = (float*)carve(4ull * Nn * HEADS);
  float*    AD2  = (float*)carve(4ull * Nn * HEADS);
  unsigned* MK2  = (unsigned*)carve(4ull * Nn * HEADS);
  float*    S2   = (float*)carve(4ull * Nn * HEADS);
  float*    EW2  = (float*)carve(4ull * E * HEADS);
  float*    AGG2 = (float*)carve(4ull * Nn * HC2);

  const int BS = 256;

  // ---- layer 1 ----
  cvt_bf16_kernel<<<blocksFor(Nn * INDIM, BS), BS, 0, stream>>>(x, XB, Nn * INDIM);
  pack_w_kernel<<<blocksFor((INDIM / 32) * (HC1 / 16) * 32, BS), BS, 0, stream>>>(
      W1, W1P, INDIM, HC1);
  pack_w_kernel<<<blocksFor((HC1 / 32) * (HC2 / 16) * 32, BS), BS, 0, stream>>>(
      W2, W2P, HC1, HC2);

  gemm_bf16_kernel<INDIM, HC1><<<blocksFor(Nn, 128), BS, 0, stream>>>(
      XB, W1P, H1, (int)Nn);

  attn_kernel<C1><<<blocksFor(Nn * HEADS, BS), BS, 0, stream>>>(
      H1, aS1, aD1, AS1, AD1, (int)Nn);

  hipMemsetAsync(MK1, 0, 4ull * Nn * HEADS, stream);
  hipMemsetAsync(S1, 0, 4ull * Nn * HEADS, stream);
  hipMemsetAsync(AGG1, 0, 4ull * Nn * HC1, stream);

  edge_max_kernel<<<blocksFor(E * HEADS, BS), BS, 0, stream>>>(
      src, dst, AS1, AD1, MK1, E * HEADS);
  edge_exp_kernel<<<blocksFor(E * HEADS, BS), BS, 0, stream>>>(
      src, dst, AS1, AD1, MK1, S1, EW1, E * HEADS);
  edge_agg_kernel<HC1, C1><<<blocksFor(E * HC1, BS), BS, 0, stream>>>(
      src, dst, EW1, H1, AGG1, E * HC1);

  fin1_kernel<<<blocksFor(Nn * HC1, BS), BS, 0, stream>>>(
      AGG1, S1, b1, XB2, Nn * HC1);

  // ---- layer 2 ----
  gemm_bf16_kernel<HC1, HC2><<<blocksFor(Nn, 128), BS, 0, stream>>>(
      XB2, W2P, H2, (int)Nn);

  attn_kernel<C2><<<blocksFor(Nn * HEADS, BS), BS, 0, stream>>>(
      H2, aS2, aD2, AS2, AD2, (int)Nn);

  hipMemsetAsync(MK2, 0, 4ull * Nn * HEADS, stream);
  hipMemsetAsync(S2, 0, 4ull * Nn * HEADS, stream);
  hipMemsetAsync(AGG2, 0, 4ull * Nn * HC2, stream);

  edge_max_kernel<<<blocksFor(E * HEADS, BS), BS, 0, stream>>>(
      src, dst, AS2, AD2, MK2, E * HEADS);
  edge_exp_kernel<<<blocksFor(E * HEADS, BS), BS, 0, stream>>>(
      src, dst, AS2, AD2, MK2, S2, EW2, E * HEADS);
  edge_agg_kernel<HC2, C2><<<blocksFor(E * HC2, BS), BS, 0, stream>>>(
      src, dst, EW2, H2, AGG2, E * HC2);

  fin2_kernel<<<blocksFor(Nn * C2, BS), BS, 0, stream>>>(
      AGG2, S2, b2, (float*)d_out, Nn * C2);
}
